// PosModel_ADS_70626442215648
// MI455X (gfx1250) — matrix operand order repaired and checked by reading, not compiled, not run on hardware
//
#include <hip/hip_runtime.h>
#include <hip/hip_bf16.h>

typedef __attribute__((ext_vector_type(16))) __bf16 v16bf;
typedef __attribute__((ext_vector_type(8)))  __bf16 v8bf;
typedef __attribute__((ext_vector_type(8)))  float  v8f;

#define BN_EPS 1e-5f

// ---------------------------------------------------------------------------
// Direct convolution (f32). Tiny channel counts (<=8) -> VALU is appropriate.
// ---------------------------------------------------------------------------
__global__ void conv2d_kernel(const float* __restrict__ X, const float* __restrict__ Wt,
                              const float* __restrict__ bias, float* __restrict__ Y,
                              int N, int Ci, int H, int W, int Co, int kh, int kw,
                              int sh, int sw, int ph, int pw, int Ho, int Wo) {
  int idx = blockIdx.x * blockDim.x + threadIdx.x;
  int total = N * Co * Ho * Wo;
  if (idx >= total) return;
  int wo = idx % Wo; int t = idx / Wo;
  int ho = t % Ho;  t /= Ho;
  int co = t % Co;  int n = t / Co;
  float acc = bias[co];
  for (int ci = 0; ci < Ci; ++ci) {
    for (int r = 0; r < kh; ++r) {
      int hi = ho * sh - ph + r;
      if (hi < 0 || hi >= H) continue;
      for (int q = 0; q < kw; ++q) {
        int wi = wo * sw - pw + q;
        if (wi < 0 || wi >= W) continue;
        acc += X[((n * Ci + ci) * H + hi) * W + wi] * Wt[((co * Ci + ci) * kh + r) * kw + q];
      }
    }
  }
  Y[idx] = acc;
}

// ---------------------------------------------------------------------------
// Batch-norm statistics: per channel c over N*HW elements of x[(n*C+c)*HW+p].
// ---------------------------------------------------------------------------
__global__ void bn_stats_kernel(const float* __restrict__ x, float* __restrict__ mean,
                                float* __restrict__ rstd, int N, int C, int HW) {
  int c = blockIdx.x;
  int tid = threadIdx.x;
  int total = N * HW;
  float s = 0.f, s2 = 0.f;
  for (int i = tid; i < total; i += blockDim.x) {
    int n = i / HW, p = i % HW;
    float v = x[(n * C + c) * HW + p];
    s += v; s2 += v * v;
  }
  __shared__ float sh[256], sh2[256];
  sh[tid] = s; sh2[tid] = s2;
  __syncthreads();
  for (int o = blockDim.x >> 1; o > 0; o >>= 1) {
    if (tid < o) { sh[tid] += sh[tid + o]; sh2[tid] += sh2[tid + o]; }
    __syncthreads();
  }
  if (tid == 0) {
    float m = sh[0] / (float)total;
    float v = sh2[0] / (float)total - m * m;
    mean[c] = m;
    rstd[c] = rsqrtf(v + BN_EPS);
  }
}

// act: 0 none, 1 relu, 2 sigmoid. Optional residual add.
// Optional bf16 mirror with padded per-sample stride ldbf (pad pre-zeroed).
__global__ void bn_apply_kernel(const float* x, const float* __restrict__ mean,
                                const float* __restrict__ rstd, const float* __restrict__ g,
                                const float* __restrict__ be, const float* res,
                                float* y, __bf16* ybf, int N, int C, int HW, int act,
                                int ldbf) {
  int idx = blockIdx.x * blockDim.x + threadIdx.x;
  int total = N * C * HW;
  if (idx >= total) return;
  int c = (idx / HW) % C;
  float v = (x[idx] - mean[c]) * rstd[c] * g[c] + be[c];
  if (res) v += res[idx];
  if (act == 1)      v = fmaxf(v, 0.f);
  else if (act == 2) v = 1.f / (1.f + __expf(-v));
  y[idx] = v;
  if (ybf) {
    int n = idx / (C * HW);
    int pos = idx % (C * HW);
    ybf[(long long)n * ldbf + pos] = (__bf16)v;
  }
}

__global__ void fill_zero_bf16_kernel(__bf16* __restrict__ p, long long n) {
  long long i = (long long)blockIdx.x * blockDim.x + threadIdx.x;
  if (i < n) p[i] = (__bf16)0.f;
}

// ---------------------------------------------------------------------------
// Pack f32 weight B[K,N] into WMMA B-fragment tiles:
// out[((tn*kBlocks + kb)*32 + lane)*16 + i] = B[kb*32 + (lane>>4)*16 + i][tn*16 + (lane&15)]
// zero-padded in K and N so the GEMM needs no guards.
// ---------------------------------------------------------------------------
__global__ void pack_b_kernel(const float* __restrict__ B, __bf16* __restrict__ out,
                              int K, int N, int tilesN, int kBlocks) {
  int idx = blockIdx.x * blockDim.x + threadIdx.x;
  int total = tilesN * kBlocks * 32 * 16;
  if (idx >= total) return;
  int i  = idx & 15;
  int l  = (idx >> 4) & 31;
  int kb = (idx >> 9) % kBlocks;
  int tn = idx / (kBlocks << 9);
  int k   = kb * 32 + (l >> 4) * 16 + i;
  int col = tn * 16 + (l & 15);
  out[idx] = (k < K && col < N) ? (__bf16)B[k * N + col] : (__bf16)0.f;
}

// ---------------------------------------------------------------------------
// bf16 WMMA GEMM, packed-B: C[M,N] = A @ B + bias.
// A: [M, lda] bf16, lda % 32 == 0, K-pad zeroed. M % 16 == 0 (true for all uses).
// Bp: fragment-packed by pack_b_kernel. One 16x16 tile per wave32.
// Inner loop: 4x 16-byte vector loads + 1 WMMA, no control flow.
// ---------------------------------------------------------------------------
__global__ void gemm_bf16_nn(const __bf16* __restrict__ A, int lda,
                             const __bf16* __restrict__ Bp, const float* __restrict__ bias,
                             float* __restrict__ C, int M, int N, int kBlocks) {
  int wavesPerBlock = blockDim.x >> 5;
  int wave = blockIdx.x * wavesPerBlock + (threadIdx.x >> 5);
  int tilesN = (N + 15) >> 4;
  int tilesM = M >> 4;
  if (wave >= tilesM * tilesN) return;          // wave-uniform: EXEC stays all-ones
  int tM = wave / tilesN, tN = wave % tilesN;
  int lane = threadIdx.x & 31;
  int half = lane >> 4, r = lane & 15;
  int arow = tM * 16 + r;
  const __bf16* aBase = A + (long long)arow * lda + half * 16;
  const __bf16* bBase = Bp + (((long long)tN * kBlocks) * 32 + lane) * 16;
  v8f c = {0.f, 0.f, 0.f, 0.f, 0.f, 0.f, 0.f, 0.f};
  for (int kb = 0; kb < kBlocks; ++kb) {
    v8bf a0 = *(const v8bf*)(aBase + kb * 32);
    v8bf a1 = *(const v8bf*)(aBase + kb * 32 + 8);
    v8bf b0 = *(const v8bf*)(bBase + (long long)kb * 512);
    v8bf b1 = *(const v8bf*)(bBase + (long long)kb * 512 + 8);
    v16bf a = __builtin_shufflevector(a0, a1, 0,1,2,3,4,5,6,7,8,9,10,11,12,13,14,15);
    v16bf b = __builtin_shufflevector(b0, b1, 0,1,2,3,4,5,6,7,8,9,10,11,12,13,14,15);
    c = __builtin_amdgcn_wmma_f32_16x16x32_bf16(false, a, false, b, (short)0, c, false, false);
  }
#pragma unroll
  for (int i = 0; i < 8; ++i) {
    int m = tM * 16 + half * 8 + i;
    int n = tN * 16 + r;
    if (n < N) C[(long long)m * N + n] = c[i] + (bias ? bias[n] : 0.f);
  }
}

// C[M,N] = A[M,lda] @ Bt[N,ldb]^T (attention scores). Strides % 32 == 0,
// K-pad zeroed. Per-lane B fragment is one contiguous 32-byte run.
__global__ void gemm_bf16_nt(const __bf16* __restrict__ A, int lda,
                             const __bf16* __restrict__ Bt, int ldb,
                             float* __restrict__ C, int M, int N, int kBlocks) {
  int wavesPerBlock = blockDim.x >> 5;
  int wave = blockIdx.x * wavesPerBlock + (threadIdx.x >> 5);
  int tilesN = (N + 15) >> 4;
  int tilesM = M >> 4;
  if (wave >= tilesM * tilesN) return;
  int tM = wave / tilesN, tN = wave % tilesN;
  int lane = threadIdx.x & 31;
  int half = lane >> 4, r = lane & 15;
  int arow = tM * 16 + r;
  int bcol = tN * 16 + r;
  const __bf16* aBase = A + (long long)arow * lda + half * 16;
  const __bf16* bBase = Bt + (long long)bcol * ldb + half * 16;
  v8f c = {0.f, 0.f, 0.f, 0.f, 0.f, 0.f, 0.f, 0.f};
  for (int kb = 0; kb < kBlocks; ++kb) {
    v8bf a0 = *(const v8bf*)(aBase + kb * 32);
    v8bf a1 = *(const v8bf*)(aBase + kb * 32 + 8);
    v8bf b0 = *(const v8bf*)(bBase + kb * 32);
    v8bf b1 = *(const v8bf*)(bBase + kb * 32 + 8);
    v16bf a = __builtin_shufflevector(a0, a1, 0,1,2,3,4,5,6,7,8,9,10,11,12,13,14,15);
    v16bf b = __builtin_shufflevector(b0, b1, 0,1,2,3,4,5,6,7,8,9,10,11,12,13,14,15);
    c = __builtin_amdgcn_wmma_f32_16x16x32_bf16(false, a, false, b, (short)0, c, false, false);
  }
#pragma unroll
  for (int i = 0; i < 8; ++i) {
    int m = tM * 16 + half * 8 + i;
    int n = tN * 16 + r;
    if (n < N) C[(long long)m * N + n] = c[i];
  }
}

// q[B,112] -> bf16 out[4][B][32] (per-head, stride padded 28->32, pad pre-zeroed)
__global__ void head_slice_cast_kernel(const float* __restrict__ q, __bf16* __restrict__ out, int B) {
  int idx = blockIdx.x * blockDim.x + threadIdx.x;
  if (idx >= B * 112) return;
  int b = idx / 112, j = idx % 112;
  int h = j / 28, d = j % 28;
  out[((long long)h * B + b) * 32 + d] = (__bf16)q[idx];
}

// fea_concat row r=b*A+a: [0:28]=inp_fea[b], [28:56]=anch_fea[a], [56:64]=0; bf16
__global__ void concat_cast_kernel(const float* __restrict__ inp, const float* __restrict__ anc,
                                   __bf16* __restrict__ out, int B, int A) {
  long long idx = (long long)blockIdx.x * blockDim.x + threadIdx.x;
  long long total = (long long)B * A * 64;
  if (idx >= total) return;
  int j = (int)(idx & 63);
  long long rr = idx >> 6;
  int a = (int)(rr % A);
  int b = (int)(rr / A);
  float v = 0.f;
  if (j < 28)      v = inp[b * 28 + j];
  else if (j < 56) v = anc[a * 28 + (j - 28)];
  out[idx] = (__bf16)v;
}

// per row b: z = max over heads(scores)*scale; sparsemax over A (<=64) anchors.
__global__ void sparsemax64_kernel(const float* __restrict__ scores, float* __restrict__ probs,
                                   int B, int A, int nheads, float scale) {
  int b = blockIdx.x;
  int t = threadIdx.x;  // 0..63
  __shared__ float zs[64], cs[64];
  __shared__ int kmax;
  float v = -1e30f;
  if (t < A) {
    for (int h = 0; h < nheads; ++h)
      v = fmaxf(v, scores[((long long)h * B + b) * A + t]);
    v *= scale;
  }
  zs[t] = v;
  if (t == 0) kmax = 0;
  __syncthreads();
  for (int k = 2; k <= 64; k <<= 1) {
    for (int j = k >> 1; j > 0; j >>= 1) {
      int ixj = t ^ j;
      if (ixj > t) {
        bool descBlk = ((t & k) == 0);
        float a0 = zs[t], a1 = zs[ixj];
        bool doSwap = descBlk ? (a0 < a1) : (a0 > a1);
        if (doSwap) { zs[t] = a1; zs[ixj] = a0; }
      }
      __syncthreads();
    }
  }
  cs[t] = zs[t];
  __syncthreads();
  for (int off = 1; off < 64; off <<= 1) {
    float add = (t >= off) ? cs[t - off] : 0.f;
    __syncthreads();
    cs[t] += add;
    __syncthreads();
  }
  bool supp = (1.f + (float)(t + 1) * zs[t] > cs[t]);
  if (supp) atomicMax(&kmax, t + 1);   // support is a prefix -> max index == count
  __syncthreads();
  if (t < A) {
    int km = kmax > 0 ? kmax : 1;
    float tau = (cs[km - 1] - 1.f) / (float)km;
    probs[(long long)b * A + t] = fmaxf(v - tau, 0.f);
  }
}

// pred[b] = sum_a (res[b,a,:] + anch_pos[a,:]) * probs[b,a]
__global__ void final_pred_kernel(const float* __restrict__ res, const float* __restrict__ anch_pos,
                                  const float* __restrict__ probs, float* __restrict__ out,
                                  int B, int A) {
  int b = blockIdx.x * blockDim.x + threadIdx.x;
  if (b >= B) return;
  float s0 = 0.f, s1 = 0.f;
  for (int a = 0; a < A; ++a) {
    float p = probs[(long long)b * A + a];
    long long r = ((long long)b * A + a) * 2;
    s0 += (res[r + 0] + anch_pos[a * 2 + 0]) * p;
    s1 += (res[r + 1] + anch_pos[a * 2 + 1]) * p;
  }
  out[b * 2 + 0] = s0;
  out[b * 2 + 1] = s1;
}

// ---------------------------------------------------------------------------
// Host orchestration
// ---------------------------------------------------------------------------
extern "C" void kernel_launch(void* const* d_in, const int* in_sizes, int n_in,
                              void* d_out, int out_size, void* d_ws, size_t ws_size,
                              hipStream_t stream) {
  (void)n_in; (void)out_size; (void)ws_size;
  enum {
    IC1 = 3, IC2 = 7, IC3 = 11,
    IB1C1 = 15, IB1C2 = 19,
    IB2C1 = 23, IB2C2 = 27, IB2EX = 31,
    IB3C1 = 35, IB3C2 = 39, IB3EX = 43,
    IBNG = 47, IBNB = 48,
    IWQ1 = 49, IWQ2 = 53, IWQ3 = 57,
    IWK1 = 61, IWK2 = 65, IWK3 = 69,
    IATG = 73, IATB = 74,
    IRE1 = 75, IRE2 = 79, IRE3W = 83, IRE3B = 84
  };
  auto P = [&](int i) -> const float* { return (const float*)d_in[i]; };

  const float* x        = (const float*)d_in[0];
  const float* anch     = (const float*)d_in[1];
  const float* anch_pos = (const float*)d_in[2];
  const int B = in_sizes[0] / (32 * 384);
  const int A = in_sizes[1] / (32 * 384);

  char* wsp = (char*)d_ws;
  size_t off = 0;
  auto alloc = [&](size_t bytes) -> void* {
    void* r = wsp + off;
    off += (bytes + 255) & ~(size_t)255;
    return r;
  };

  float* s_mean = (float*)alloc(sizeof(float) * 256);
  float* s_rstd = (float*)alloc(sizeof(float) * 256);

  auto conv = [&](const float* X, int N, int Ci, int H, int W, int Co, int kh, int kw,
                  int sh, int sw, int ph, int pw, const float* Wt, const float* bias,
                  float* Y, int Ho, int Wo) {
    int total = N * Co * Ho * Wo;
    conv2d_kernel<<<(total + 255) / 256, 256, 0, stream>>>(X, Wt, bias, Y, N, Ci, H, W,
                                                           Co, kh, kw, sh, sw, ph, pw, Ho, Wo);
  };
  auto bn = [&](float* xin, int N, int C, int HW, const float* g, const float* be,
                const float* res, float* y, __bf16* ybf, int act, int ldbf) {
    bn_stats_kernel<<<C, 256, 0, stream>>>(xin, s_mean, s_rstd, N, C, HW);
    int total = N * C * HW;
    bn_apply_kernel<<<(total + 255) / 256, 256, 0, stream>>>(xin, s_mean, s_rstd, g, be,
                                                             res, y, ybf, N, C, HW, act, ldbf);
  };
  auto zero_bf = [&](__bf16* p, long long n) {
    fill_zero_bf16_kernel<<<(int)((n + 255) / 256), 256, 0, stream>>>(p, n);
  };
  // pack f32 weight [K,N] into WMMA fragment tiles; returns kBlocks via ref
  auto pack_w = [&](const float* Wf, int K, int N, __bf16** outp, int* kBlocksOut) {
    int kBlocks = (K + 31) / 32;
    int tilesN = (N + 15) / 16;
    int total = tilesN * kBlocks * 512;
    __bf16* Bp = (__bf16*)alloc(2 * (size_t)total);
    pack_b_kernel<<<(total + 255) / 256, 256, 0, stream>>>(Wf, Bp, K, N, tilesN, kBlocks);
    *outp = Bp;
    *kBlocksOut = kBlocks;
  };
  auto gemm = [&](const __bf16* Am, int lda, const __bf16* Bp, int kBlocks,
                  const float* bias, float* Cm, int M, int N) {
    int tiles = (M / 16) * ((N + 15) / 16);
    gemm_bf16_nn<<<(tiles + 7) / 8, 256, 0, stream>>>(Am, lda, Bp, bias, Cm, M, N, kBlocks);
  };

  // ---- feature extractor (batch-stat BN => stats pass per stage) ----
  auto fea_extract = [&](const float* X, int N, float* fea, __bf16* feab /*[N,32] pre-zeroed*/) {
    float* t1  = (float*)alloc(sizeof(float) * (size_t)N * 8 * 31 * 47);
    float* t2  = (float*)alloc(sizeof(float) * (size_t)N * 8 * 10 * 15);
    float* t3  = (float*)alloc(sizeof(float) * (size_t)N * 8 * 4 * 7);
    float* r1a = (float*)alloc(sizeof(float) * (size_t)N * 8 * 28);
    float* r1b = (float*)alloc(sizeof(float) * (size_t)N * 8 * 28);
    float* r1o = (float*)alloc(sizeof(float) * (size_t)N * 8 * 28);
    float* r2a = (float*)alloc(sizeof(float) * (size_t)N * 6 * 28);
    float* r2b = (float*)alloc(sizeof(float) * (size_t)N * 6 * 28);
    float* r2e = (float*)alloc(sizeof(float) * (size_t)N * 6 * 28);
    float* r2o = (float*)alloc(sizeof(float) * (size_t)N * 6 * 28);
    float* r3a = (float*)alloc(sizeof(float) * (size_t)N * 28);
    float* r3b = (float*)alloc(sizeof(float) * (size_t)N * 28);
    float* r3e = (float*)alloc(sizeof(float) * (size_t)N * 28);
    float* r3o = (float*)alloc(sizeof(float) * (size_t)N * 28);

    conv(X, N, 1, 32, 384, 8, 2, 9, 1, 8, 0, 0, P(IC1), P(IC1 + 1), t1, 31, 47);
    bn(t1, N, 8, 31 * 47, P(IC1 + 2), P(IC1 + 3), nullptr, t1, nullptr, 1, 0);
    conv(t1, N, 8, 31, 47, 8, 3, 4, 3, 3, 0, 0, P(IC2), P(IC2 + 1), t2, 10, 15);
    bn(t2, N, 8, 10 * 15, P(IC2 + 2), P(IC2 + 3), nullptr, t2, nullptr, 1, 0);
    conv(t2, N, 8, 10, 15, 8, 3, 3, 2, 2, 0, 0, P(IC3), P(IC3 + 1), t3, 4, 7);
    bn(t3, N, 8, 4 * 7, P(IC3 + 2), P(IC3 + 3), nullptr, t3, nullptr, 1, 0);
    // blk1 (identity shortcut)
    conv(t3, N, 8, 4, 7, 8, 3, 3, 1, 1, 1, 1, P(IB1C1), P(IB1C1 + 1), r1a, 4, 7);
    bn(r1a, N, 8, 28, P(IB1C1 + 2), P(IB1C1 + 3), nullptr, r1a, nullptr, 1, 0);
    conv(r1a, N, 8, 4, 7, 8, 3, 3, 1, 1, 1, 1, P(IB1C2), P(IB1C2 + 1), r1b, 4, 7);
    bn(r1b, N, 8, 28, P(IB1C2 + 2), P(IB1C2 + 3), t3, r1o, nullptr, 1, 0);
    // blk2 (1x1 shortcut 8->6)
    conv(r1o, N, 8, 4, 7, 6, 3, 3, 1, 1, 1, 1, P(IB2C1), P(IB2C1 + 1), r2a, 4, 7);
    bn(r2a, N, 6, 28, P(IB2C1 + 2), P(IB2C1 + 3), nullptr, r2a, nullptr, 1, 0);
    conv(r2a, N, 6, 4, 7, 6, 3, 3, 1, 1, 1, 1, P(IB2C2), P(IB2C2 + 1), r2b, 4, 7);
    conv(r1o, N, 8, 4, 7, 6, 1, 1, 1, 1, 0, 0, P(IB2EX), P(IB2EX + 1), r2e, 4, 7);
    bn(r2e, N, 6, 28, P(IB2EX + 2), P(IB2EX + 3), nullptr, r2e, nullptr, 0, 0);
    bn(r2b, N, 6, 28, P(IB2C2 + 2), P(IB2C2 + 3), r2e, r2o, nullptr, 1, 0);
    // blk3 (1x1 shortcut 6->1)
    conv(r2o, N, 6, 4, 7, 1, 3, 3, 1, 1, 1, 1, P(IB3C1), P(IB3C1 + 1), r3a, 4, 7);
    bn(r3a, N, 1, 28, P(IB3C1 + 2), P(IB3C1 + 3), nullptr, r3a, nullptr, 1, 0);
    conv(r3a, N, 1, 4, 7, 1, 3, 3, 1, 1, 1, 1, P(IB3C2), P(IB3C2 + 1), r3b, 4, 7);
    conv(r2o, N, 6, 4, 7, 1, 1, 1, 1, 1, 0, 0, P(IB3EX), P(IB3EX + 1), r3e, 4, 7);
    bn(r3e, N, 1, 28, P(IB3EX + 2), P(IB3EX + 3), nullptr, r3e, nullptr, 0, 0);
    bn(r3b, N, 1, 28, P(IB3C2 + 2), P(IB3C2 + 3), r3e, r3o, nullptr, 1, 0);
    // bn1d over flattened [N,28]; bf16 mirror with K padded to 32
    bn(r3o, N, 28, 1, P(IBNG), P(IBNB), nullptr, fea, feab, 0, 32);
  };

  float*  anch_fea  = (float*)alloc(sizeof(float) * (size_t)A * 28);
  __bf16* anch_feab = (__bf16*)alloc(2 * (size_t)A * 32);
  float*  inp_fea   = (float*)alloc(sizeof(float) * (size_t)B * 28);
  __bf16* inp_feab  = (__bf16*)alloc(2 * (size_t)B * 32);
  zero_bf(anch_feab, (long long)A * 32);
  zero_bf(inp_feab, (long long)B * 32);
  fea_extract(anch, A, anch_fea, anch_feab);
  fea_extract(x, B, inp_fea, inp_feab);

  // ---- Q/K MLPs via bf16 WMMA GEMM (packed weights) ----
  auto mlp_qk = [&](const __bf16* feab, int N, int L1, int L2, int L3, float* outp) {
    __bf16 *w1, *w2, *w3;
    int kb1, kb2, kb3;
    pack_w(P(L1), 28, 224, &w1, &kb1);
    pack_w(P(L2), 224, 224, &w2, &kb2);
    pack_w(P(L3), 224, 112, &w3, &kb3);
    float*  h1  = (float*)alloc(sizeof(float) * (size_t)N * 224);
    __bf16* h1b = (__bf16*)alloc(2 * (size_t)N * 224);
    float*  h2  = (float*)alloc(sizeof(float) * (size_t)N * 224);
    __bf16* h2b = (__bf16*)alloc(2 * (size_t)N * 224);
    gemm(feab, 32, w1, kb1, P(L1 + 1), h1, N, 224);
    bn(h1, N, 224, 1, P(L1 + 2), P(L1 + 3), nullptr, h1, h1b, 1, 224);
    gemm(h1b, 224, w2, kb2, P(L2 + 1), h2, N, 224);
    bn(h2, N, 224, 1, P(L2 + 2), P(L2 + 3), nullptr, h2, h2b, 1, 224);
    gemm(h2b, 224, w3, kb3, P(L3 + 1), outp, N, 112);
    bn(outp, N, 112, 1, P(L3 + 2), P(L3 + 3), nullptr, outp, nullptr, 2, 0);
  };

  float* q  = (float*)alloc(sizeof(float) * (size_t)B * 112);
  float* kk = (float*)alloc(sizeof(float) * (size_t)A * 112);
  mlp_qk(inp_feab, B, IWQ1, IWQ2, IWQ3, q);
  mlp_qk(anch_feab, A, IWK1, IWK2, IWK3, kk);

  // per-head BN (batch stats, separately for Q and K), then per-head bf16 pack
  bn(q, B, 4, 28, P(IATG), P(IATB), nullptr, q, nullptr, 0, 0);
  bn(kk, A, 4, 28, P(IATG), P(IATB), nullptr, kk, nullptr, 0, 0);
  __bf16* qb = (__bf16*)alloc(2 * (size_t)4 * B * 32);
  __bf16* kb = (__bf16*)alloc(2 * (size_t)4 * A * 32);
  zero_bf(qb, (long long)4 * B * 32);
  zero_bf(kb, (long long)4 * A * 32);
  head_slice_cast_kernel<<<(B * 112 + 255) / 256, 256, 0, stream>>>(q, qb, B);
  head_slice_cast_kernel<<<(A * 112 + 255) / 256, 256, 0, stream>>>(kk, kb, A);

  // scores[h] = Qh @ Kh^T  (WMMA NT GEMM per head, single 32-wide k-block)
  float* scores = (float*)alloc(sizeof(float) * (size_t)4 * B * A);
  for (int h = 0; h < 4; ++h) {
    int tiles = (B / 16) * ((A + 15) / 16);
    gemm_bf16_nt<<<(tiles + 7) / 8, 256, 0, stream>>>(qb + (size_t)h * B * 32, 32,
                                                      kb + (size_t)h * A * 32, 32,
                                                      scores + (size_t)h * B * A, B, A, 1);
  }
  float* probs = (float*)alloc(sizeof(float) * (size_t)B * A);
  sparsemax64_kernel<<<B, 64, 0, stream>>>(scores, probs, B, A, 4, 1.f / sqrtf(28.f));

  // ---- pair regression MLP on [B*A, 56->64] via WMMA ----
  long long M = (long long)B * A;
  __bf16* cc = (__bf16*)alloc(2 * (size_t)M * 64);
  concat_cast_kernel<<<(int)((M * 64 + 255) / 256), 256, 0, stream>>>(inp_fea, anch_fea, cc, B, A);
  __bf16 *rw1, *rw2, *rw3;
  int rkb1, rkb2, rkb3;
  pack_w(P(IRE1), 56, 16, &rw1, &rkb1);
  pack_w(P(IRE2), 16, 8, &rw2, &rkb2);
  pack_w(P(IRE3W), 8, 2, &rw3, &rkb3);
  float*  rh1  = (float*)alloc(sizeof(float) * (size_t)M * 16);
  __bf16* rh1b = (__bf16*)alloc(2 * (size_t)M * 32);
  float*  rh2  = (float*)alloc(sizeof(float) * (size_t)M * 8);
  __bf16* rh2b = (__bf16*)alloc(2 * (size_t)M * 32);
  float*  res  = (float*)alloc(sizeof(float) * (size_t)M * 2);
  zero_bf(rh1b, M * 32);
  zero_bf(rh2b, M * 32);
  gemm(cc, 64, rw1, rkb1, P(IRE1 + 1), rh1, (int)M, 16);
  bn(rh1, (int)M, 16, 1, P(IRE1 + 2), P(IRE1 + 3), nullptr, rh1, rh1b, 1, 32);
  gemm(rh1b, 32, rw2, rkb2, P(IRE2 + 1), rh2, (int)M, 8);
  bn(rh2, (int)M, 8, 1, P(IRE2 + 2), P(IRE2 + 3), nullptr, rh2, rh2b, 1, 32);
  gemm(rh2b, 32, rw3, rkb3, P(IRE3B), res, (int)M, 2);

  final_pred_kernel<<<(B + 255) / 256, 256, 0, stream>>>(res, anch_pos, probs, (float*)d_out, B, A);
}
